// PhysicsAwareAttention_78134045049024
// MI455X (gfx1250) — compile-verified, matrix operand order checked by reading, not run on hardware
//
#include <hip/hip_runtime.h>

#define NN    40000
#define EE    640000
#define HIDD  128
#define NHH   8
#define HDD   16
#define SBLK  20000   /* (EE*NHH)/256 score blocks */

typedef __attribute__((ext_vector_type(8)))  float        v8f;
typedef __attribute__((ext_vector_type(8)))  unsigned int v8u;
typedef __attribute__((ext_vector_type(16))) __bf16       v16bf;

// ---------------- bf16 helpers (raw-bit, RNE) ----------------
__device__ __forceinline__ float bf_lo(unsigned int u) { return __uint_as_float(u << 16); }
__device__ __forceinline__ float bf_hi(unsigned int u) { return __uint_as_float(u & 0xffff0000u); }
__device__ __forceinline__ unsigned int f2bf_rne(float f) {
    unsigned int u = __float_as_uint(f);
    return (u + 0x7fffu + ((u >> 16) & 1u)) >> 16;
}
__device__ __forceinline__ unsigned int pack2bf(float a, float b) {
    return f2bf_rne(a) | (f2bf_rne(b) << 16);
}

// ---------------- tiny kernels ----------------
__global__ void k_field_embed(const float* __restrict__ fv, const float* __restrict__ Wf,
                              const float* __restrict__ bfv, float* __restrict__ fe) {
    int b = blockIdx.x, j = threadIdx.x;           // 64 blocks x 128 threads
    float v = bfv[j];
    v += fv[b * 3 + 0] * Wf[0 * HIDD + j];
    v += fv[b * 3 + 1] * Wf[1 * HIDD + j];
    v += fv[b * 3 + 2] * Wf[2 * HIDD + j];
    fe[b * HIDD + j] = v;
}

// pack Wq/Wk/Wv/Wo (row-major [K][N] fp32) into WMMA B-fragment order.
// Layout: idx = ((((w*8 + nt)*4 + kb)*32 + lane)*8 + vgpr)  -> each lane's 8 dwords are
// CONTIGUOUS (32B), so the GEMM mainloop loads one fragment as two global_load_b128.
__global__ void k_pack_w(const float* __restrict__ Wq, const float* __restrict__ Wk,
                         const float* __restrict__ Wv, const float* __restrict__ Wo,
                         unsigned int* __restrict__ bpk) {
    int t    = blockIdx.x * 256 + threadIdx.x;     // 0..32767
    int v    = t & 7;
    int lane = (t >> 3) & 31;
    int kb   = (t >> 8) & 3;
    int nt   = (t >> 10) & 7;
    int w    = t >> 13;
    const float* W = (w == 0) ? Wq : (w == 1) ? Wk : (w == 2) ? Wv : Wo;
    int nl    = lane & 15;
    int khalf = lane >> 4;
    int kl    = ((v < 4) ? 0 : 16) + 8 * khalf + 2 * (v & 3);
    int k     = kb * 32 + kl;
    int n     = nt * 16 + nl;
    bpk[t] = pack2bf(W[(size_t)k * HIDD + n], W[(size_t)(k + 1) * HIDD + n]);
}

__global__ void k_zero4(float4* __restrict__ p) {
    p[(size_t)blockIdx.x * 256 + threadIdx.x] = make_float4(0.f, 0.f, 0.f, 0.f);
}

// ---------------- WMMA GEMM core ----------------
// Stage a 64x128 fp32 block into LDS as bf16 pairs, row stride 68 dwords (bank-conflict-free).
__device__ __forceinline__ void stage_a(unsigned int* lds, const float* __restrict__ A, int rowBase) {
    for (int idx = threadIdx.x; idx < 64 * 64; idx += 128) {
        int row = idx >> 6, pc = idx & 63;
        const float* ap = A + (size_t)(rowBase + row) * HIDD + pc * 2;
        lds[row * 68 + pc] = pack2bf(ap[0], ap[1]);
    }
}

// One wave computes a 16-row x 128-col tile: acc[nt] for nt=0..7, K=128 in 4 steps of 32.
__device__ __forceinline__ void wmma_mainloop(const unsigned int* lds, int wave, int lane,
                                              const unsigned int* __restrict__ bw, v8f acc[8]) {
    const int khalf = lane >> 4;
    const int r     = wave * 16 + (lane & 15);
#pragma unroll
    for (int kb = 0; kb < 4; ++kb) {
        const unsigned int* ar = lds + r * 68 + kb * 16 + 4 * khalf;
        v8u ap;
        ap[0] = ar[0]; ap[1] = ar[1]; ap[2] = ar[2]; ap[3] = ar[3];
        ap[4] = ar[8]; ap[5] = ar[9]; ap[6] = ar[10]; ap[7] = ar[11];
        v16bf afrag = __builtin_bit_cast(v16bf, ap);
#pragma unroll
        for (int nt = 0; nt < 8; ++nt) {
            // 32B contiguous per lane -> two global_load_b128, coalesced across lanes
            v8u bp = *(const v8u*)(bw + (((nt * 4 + kb) * 32) + lane) * 8);
            v16bf bfrag = __builtin_bit_cast(v16bf, bp);
            acc[nt] = __builtin_amdgcn_wmma_f32_16x16x32_bf16(
                false, afrag, false, bfrag, (short)0, acc[nt], false, false);
        }
    }
}

// Fused Q/K/V projection: x(fp32) -> LDS(bf16) once, 3 weight passes, bf16 outputs.
__global__ void __launch_bounds__(128)
k_gemm_qkv(const float* __restrict__ x, const unsigned int* __restrict__ bpk,
           const float* __restrict__ bq, const float* __restrict__ bk, const float* __restrict__ bv,
           const float* __restrict__ fe, const int* __restrict__ batch,
           unsigned short* __restrict__ Qb, unsigned short* __restrict__ Kb,
           unsigned short* __restrict__ Vb) {
    __shared__ unsigned int lds[64 * 68];
    const int rowBase = blockIdx.x * 64;
    stage_a(lds, x, rowBase);
    __syncthreads();
    const int wave = threadIdx.x >> 5, lane = threadIdx.x & 31;
    const int colb = lane & 15, half = lane >> 4;
    for (int w = 0; w < 3; ++w) {
        v8f acc[8] = {};
        wmma_mainloop(lds, wave, lane, bpk + w * 8192, acc);
        const float* bias = (w == 0) ? bq : (w == 1) ? bk : bv;
        unsigned short* dst = (w == 0) ? Qb : (w == 1) ? Kb : Vb;
#pragma unroll
        for (int nt = 0; nt < 8; ++nt) {
            int col = nt * 16 + colb;
            float bb = bias[col];
#pragma unroll
            for (int j = 0; j < 8; ++j) {
                int row = rowBase + wave * 16 + j + 8 * half;
                float v = acc[nt][j] + bb;
                if (w == 1) v += fe[(size_t)batch[row] * HIDD + col];
                dst[(size_t)row * HIDD + col] = (unsigned short)f2bf_rne(v);
            }
        }
    }
}

// Output projection: agg(fp32) @ Wo + bo + x -> fp32 out
__global__ void __launch_bounds__(128)
k_gemm_out(const float* __restrict__ agg, const unsigned int* __restrict__ bpk,
           const float* __restrict__ bo, const float* __restrict__ x, float* __restrict__ out) {
    __shared__ unsigned int lds[64 * 68];
    const int rowBase = blockIdx.x * 64;
    stage_a(lds, agg, rowBase);
    __syncthreads();
    const int wave = threadIdx.x >> 5, lane = threadIdx.x & 31;
    v8f acc[8] = {};
    wmma_mainloop(lds, wave, lane, bpk + 3 * 8192, acc);
    const int colb = lane & 15, half = lane >> 4;
#pragma unroll
    for (int nt = 0; nt < 8; ++nt) {
        int col = nt * 16 + colb;
        float bb = bo[col];
#pragma unroll
        for (int j = 0; j < 8; ++j) {
            int row = rowBase + wave * 16 + j + 8 * half;
            out[(size_t)row * HIDD + col] = acc[nt][j] + bb + x[(size_t)row * HIDD + col];
        }
    }
}

// ---------------- edge phase ----------------
// one thread per (edge, head): 16-dim bf16 dot, + per-block per-head max partials
__global__ void k_edge_scores(const unsigned short* __restrict__ Qb, const unsigned short* __restrict__ Kb,
                              const int* __restrict__ ei, float* __restrict__ scores,
                              float* __restrict__ partial) {
    int t = blockIdx.x * 256 + threadIdx.x;
    int e = t >> 3, h = t & 7;
    int src = ei[e], dst = ei[EE + e];
    v8u q = *(const v8u*)(Qb + (size_t)src * HIDD + h * HDD);
    v8u k = *(const v8u*)(Kb + (size_t)dst * HIDD + h * HDD);
    float s = 0.f;
#pragma unroll
    for (int i = 0; i < 8; ++i) s += bf_lo(q[i]) * bf_lo(k[i]) + bf_hi(q[i]) * bf_hi(k[i]);
    s *= 0.25f;  // 1/sqrt(HEAD_DIM)
    scores[t] = s;
    __shared__ float red[256];
    red[threadIdx.x] = s;
    __syncthreads();
    for (int st = 128; st >= 8; st >>= 1) {       // strides are multiples of 8 -> per-head
        if (threadIdx.x < st) red[threadIdx.x] = fmaxf(red[threadIdx.x], red[threadIdx.x + st]);
        __syncthreads();
    }
    if (threadIdx.x < 8) partial[(size_t)blockIdx.x * 8 + threadIdx.x] = red[threadIdx.x];
}

__global__ void k_reduce_max(const float* __restrict__ partial, float* __restrict__ headv) {
    int h = blockIdx.x;                            // 8 blocks
    float m = -3.4e38f;
    for (int i = threadIdx.x; i < SBLK; i += 256) m = fmaxf(m, partial[(size_t)i * 8 + h]);
    __shared__ float red[256];
    red[threadIdx.x] = m;
    __syncthreads();
    for (int st = 128; st >= 1; st >>= 1) {
        if (threadIdx.x < st) red[threadIdx.x] = fmaxf(red[threadIdx.x], red[threadIdx.x + st]);
        __syncthreads();
    }
    if (threadIdx.x == 0) headv[h] = red[0];
}

__global__ void k_exp_sum(float* __restrict__ scores, const float* __restrict__ headmax,
                          float* __restrict__ partial) {
    int t = blockIdx.x * 256 + threadIdx.x;
    int h = t & 7;
    float p = __expf(scores[t] - headmax[h]);
    scores[t] = p;
    __shared__ float red[256];
    red[threadIdx.x] = p;
    __syncthreads();
    for (int st = 128; st >= 8; st >>= 1) {
        if (threadIdx.x < st) red[threadIdx.x] += red[threadIdx.x + st];
        __syncthreads();
    }
    if (threadIdx.x < 8) partial[(size_t)blockIdx.x * 8 + threadIdx.x] = red[threadIdx.x];
}

__global__ void k_reduce_sum(const float* __restrict__ partial, float* __restrict__ headv) {
    int h = blockIdx.x;
    float s = 0.f;
    for (int i = threadIdx.x; i < SBLK; i += 256) s += partial[(size_t)i * 8 + h];
    __shared__ float red[256];
    red[threadIdx.x] = s;
    __syncthreads();
    for (int st = 128; st >= 1; st >>= 1) {
        if (threadIdx.x < st) red[threadIdx.x] += red[threadIdx.x + st];
        __syncthreads();
    }
    if (threadIdx.x == 0) headv[h] = red[0];
}

// 32 threads per edge, 4 cols each: attn * V[dst] scatter-added into agg[src]
__global__ void k_scatter(const float* __restrict__ scores, const float* __restrict__ headsum,
                          const unsigned short* __restrict__ Vb, const int* __restrict__ ei,
                          float* __restrict__ agg) {
    size_t t = (size_t)blockIdx.x * 256 + threadIdx.x;
    int e = (int)(t >> 5), j = (int)(t & 31);
    int h = j >> 2;
    float p = scores[(size_t)e * 8 + h] / headsum[h];
    int src = ei[e], dst = ei[EE + e];
    const unsigned int* vp = (const unsigned int*)(Vb + (size_t)dst * HIDD + j * 4);
    unsigned int v0 = vp[0], v1 = vp[1];
    float* ag = agg + (size_t)src * HIDD + j * 4;
    atomicAdd(ag + 0, p * bf_lo(v0));
    atomicAdd(ag + 1, p * bf_hi(v0));
    atomicAdd(ag + 2, p * bf_lo(v1));
    atomicAdd(ag + 3, p * bf_hi(v1));
}

// ---------------- launch ----------------
extern "C" void kernel_launch(void* const* d_in, const int* in_sizes, int n_in,
                              void* d_out, int out_size, void* d_ws, size_t ws_size,
                              hipStream_t stream) {
    const float* x    = (const float*)d_in[0];
    const int*   ei   = (const int*)d_in[1];
    /* d_in[2] = edge_attr: unused by the reference */
    const float* fv   = (const float*)d_in[3];
    const int*   bat  = (const int*)d_in[4];
    const float* Wq   = (const float*)d_in[5];
    const float* bq   = (const float*)d_in[6];
    const float* Wk   = (const float*)d_in[7];
    const float* bk   = (const float*)d_in[8];
    const float* Wv   = (const float*)d_in[9];
    const float* bv   = (const float*)d_in[10];
    const float* Wf   = (const float*)d_in[11];
    const float* bfv  = (const float*)d_in[12];
    const float* Wo   = (const float*)d_in[13];
    const float* bo   = (const float*)d_in[14];
    float*       out  = (float*)d_out;

    // workspace layout (all section sizes are multiples of 256B)
    const size_t SZ_QKV  = (size_t)NN * HIDD * 2;   // bf16 Q/K/V: 10,240,000 each
    const size_t SZ_SC   = (size_t)EE * NHH * 4;    // scores: 20,480,000
    const size_t SZ_AGG  = (size_t)NN * HIDD * 4;   // agg: 20,480,000
    char* wsb = (char*)d_ws;
    unsigned short* Qb      = (unsigned short*)(wsb);
    unsigned short* Kb      = (unsigned short*)(wsb + SZ_QKV);
    unsigned short* Vb      = (unsigned short*)(wsb + 2 * SZ_QKV);
    float*          scores  = (float*)(wsb + 3 * SZ_QKV);
    float*          agg     = (float*)(wsb + 3 * SZ_QKV + SZ_SC);
    float*          partial = (float*)(wsb + 3 * SZ_QKV + SZ_SC + SZ_AGG);
    float*          headmax = partial + SBLK * 8;
    float*          headsum = headmax + 64;
    float*          fe      = headsum + 64;
    unsigned int*   bpk     = (unsigned int*)(fe + 64 * HIDD);

    k_field_embed<<<64, 128, 0, stream>>>(fv, Wf, bfv, fe);
    k_pack_w<<<128, 256, 0, stream>>>(Wq, Wk, Wv, Wo, bpk);
    k_zero4<<<(NN * HIDD / 4) / 256, 256, 0, stream>>>((float4*)agg);
    k_gemm_qkv<<<NN / 64, 128, 0, stream>>>(x, bpk, bq, bk, bv, fe, bat, Qb, Kb, Vb);
    k_edge_scores<<<SBLK, 256, 0, stream>>>(Qb, Kb, ei, scores, partial);
    k_reduce_max<<<8, 256, 0, stream>>>(partial, headmax);
    k_exp_sum<<<SBLK, 256, 0, stream>>>(scores, headmax, partial);
    k_reduce_sum<<<8, 256, 0, stream>>>(partial, headsum);
    k_scatter<<<(unsigned)((size_t)EE * 32 / 256), 256, 0, stream>>>(scores, headsum, Vb, ei, agg);
    k_gemm_out<<<NN / 64, 128, 0, stream>>>(agg, bpk, bo, x, out);
}
